// SR_Model_72825465471064
// MI455X (gfx1250) — compile-verified, matrix operand order checked by reading, not run on hardware
//
#include <hip/hip_runtime.h>

typedef __attribute__((ext_vector_type(16))) _Float16 v16h;
typedef __attribute__((ext_vector_type(8)))  _Float16 v8h;
typedef __attribute__((ext_vector_type(8)))  float    v8f;

union V16 { v16h v; v8h h[2]; };

#define HB 360
#define WB 640
#define HT 72
#define WT 128
#define NPATCH (2*HT*WT)     // 18432
#define XSTRIPS (WB/16)      // 40
#define NSTRIP (2*HB*XSTRIPS) // 28800

// K-offset for the CDNA5 16-bit A/B fragment interleave:
// lane<16 holds K {0..7,16..23}; lane>=16 holds K {8..15,24..31}.
__device__ __forceinline__ int kmap(int ll, int e) {
    return e + (e < 8 ? 0 : 8) + ((ll >= 16) ? 8 : 0);
}

__device__ __forceinline__ void load_frag(const _Float16* p, V16& f) {
    const v8h* q = (const v8h*)p;
    f.h[0] = q[0]; f.h[1] = q[2];
}

// ---------------- stage 0: im2col for feat conv -----------------------------
// rows = NPATCH*9 (patch, 3x3 out pos), K = 27 padded to 32, k = c*9+kh*3+kw
__global__ void k_im2col1(const float* __restrict__ x, _Float16* __restrict__ A1) {
    int row = blockIdx.x * blockDim.x + threadIdx.x;
    if (row >= NPATCH * 9) return;
    int p = row / 9, pos = row % 9;
    int oy = pos / 3, ox = pos % 3;
    int b = p / (HT*WT); int rem = p % (HT*WT);
    int ty = rem / WT, tx = rem % WT;
    _Float16* dst = A1 + (size_t)row * 32;
    #pragma unroll
    for (int c = 0; c < 3; ++c)
        #pragma unroll
        for (int kh = 0; kh < 3; ++kh)
            #pragma unroll
            for (int kw = 0; kw < 3; ++kw) {
                float v = x[(((size_t)b*3 + c)*HB + ty*5 + oy + kh)*WB + tx*5 + ox + kw];
                dst[c*9 + kh*3 + kw] = (_Float16)v;
            }
    #pragma unroll
    for (int k = 27; k < 32; ++k) dst[k] = (_Float16)0.f;
}

// ---------------- weight pack kernels (f32 -> f16, WMMA-friendly) -----------
__global__ void k_packB1(const float* __restrict__ w, _Float16* __restrict__ B) {
    int i = blockIdx.x * blockDim.x + threadIdx.x;
    if (i >= 256*32) return;
    int o = i >> 5, k = i & 31;
    B[i] = (_Float16)(k < 27 ? w[o*27 + k] : 0.f);
}
// Bk[o][pos*256 + c] = w[(o*256 + c)*9 + pos]  (K order matches feat buffer)
__global__ void k_packB29(const float* __restrict__ w, _Float16* __restrict__ B, int Nout) {
    int i = blockIdx.x * blockDim.x + threadIdx.x;
    if (i >= Nout * 2304) return;
    int o = i / 2304, kk = i % 2304;
    int pos = kk >> 8, c = kk & 255;
    B[i] = (_Float16)w[((size_t)o*256 + c)*9 + pos];
}
// conv1 B fragments: [(tap*8 + ntile)*32 + lane]*16 halfs, K = in-channel (32)
__global__ void k_packW1(const float* __restrict__ w, _Float16* __restrict__ W) {
    int i = blockIdx.x * blockDim.x + threadIdx.x;
    if (i >= 25*8*32*16) return;
    int e = i & 15, ll = (i >> 4) & 31, j = (i >> 9) & 7, t = i >> 12;
    int n = j*16 + (ll & 15);
    int k = kmap(ll, e);
    W[i] = (_Float16)w[((size_t)n*32 + k)*25 + t];   // w1 OIHW (128,32,5,5)
}
// conv2: frag = ((tap*4 + kchunk)*2 + ntile)
__global__ void k_packW2(const float* __restrict__ w, _Float16* __restrict__ W) {
    int i = blockIdx.x * blockDim.x + threadIdx.x;
    if (i >= 9*4*2*32*16) return;
    int e = i & 15, ll = (i >> 4) & 31, j = (i >> 9) & 1, kc = (i >> 10) & 3, t = i >> 12;
    int n = j*16 + (ll & 15);
    int k = kc*32 + kmap(ll, e);
    W[i] = (_Float16)w[((size_t)n*128 + k)*9 + t];   // w2 OIHW (32,128,3,3)
}
// conv3: frag = tap*3 + ntile
__global__ void k_packW3(const float* __restrict__ w, _Float16* __restrict__ W) {
    int i = blockIdx.x * blockDim.x + threadIdx.x;
    if (i >= 9*3*32*16) return;
    int e = i & 15, ll = (i >> 4) & 31, f = i >> 9;
    int j = f % 3, t = f / 3;
    int n = j*16 + (ll & 15);
    int k = kmap(ll, e);
    W[i] = (_Float16)w[((size_t)n*32 + k)*9 + t];    // w3 OIHW (48,32,3,3)
}

// ---------------- generic WMMA GEMM: C[M,N] = A[M,K] * B[N,K]^T + bias ------
// one wave -> 32(M) x 32(N) register tile: 4 WMMA per 8 fragment loads,
// halving L2 traffic per FLOP vs a 16x32 tile.
__global__ void k_gemm_f16(const _Float16* __restrict__ A, const _Float16* __restrict__ B,
                           const float* __restrict__ bias, _Float16* __restrict__ C,
                           int mtiles2, int npairs, int K, int relu) {
    int lane = threadIdx.x & 31;
    int wid  = blockIdx.x * (blockDim.x >> 5) + (threadIdx.x >> 5);
    if (wid >= mtiles2 * npairs) return;
    int mt = wid / npairs, np = wid % npairs;
    int lm = lane & 15, off = (lane >> 4) * 8;
    size_t arow0 = (size_t)(mt*32 + lm) * K;
    size_t arow1 = (size_t)(mt*32 + 16 + lm) * K;
    size_t brow0 = (size_t)(np*32 + lm) * K;
    size_t brow1 = (size_t)(np*32 + 16 + lm) * K;
    v8f z = {};
    v8f acc00 = z, acc01 = z, acc10 = z, acc11 = z;
    for (int k0 = 0; k0 < K; k0 += 32) {
        __builtin_prefetch(A + arow0 + k0 + 256, 0, 1);
        __builtin_prefetch(A + arow1 + k0 + 256, 0, 1);
        V16 a0, a1, b0, b1;
        load_frag(A + arow0 + k0 + off, a0);
        load_frag(A + arow1 + k0 + off, a1);
        load_frag(B + brow0 + k0 + off, b0);
        load_frag(B + brow1 + k0 + off, b1);
        acc00 = __builtin_amdgcn_wmma_f32_16x16x32_f16(false, a0.v, false, b0.v, (short)0, acc00, false, false);
        acc01 = __builtin_amdgcn_wmma_f32_16x16x32_f16(false, a0.v, false, b1.v, (short)0, acc01, false, false);
        acc10 = __builtin_amdgcn_wmma_f32_16x16x32_f16(false, a1.v, false, b0.v, (short)0, acc10, false, false);
        acc11 = __builtin_amdgcn_wmma_f32_16x16x32_f16(false, a1.v, false, b1.v, (short)0, acc11, false, false);
    }
    int n0 = np*32 + lm, n1 = n0 + 16;
    int N = npairs * 32;
    float bb0 = bias[n0], bb1 = bias[n1];
    #pragma unroll
    for (int v = 0; v < 8; ++v) {
        int m0 = mt*32 + v + 8*(lane >> 4);
        int m1 = m0 + 16;
        float x00 = acc00[v] + bb0, x01 = acc01[v] + bb1;
        float x10 = acc10[v] + bb0, x11 = acc11[v] + bb1;
        if (relu) {
            x00 = fmaxf(x00, 0.f); x01 = fmaxf(x01, 0.f);
            x10 = fmaxf(x10, 0.f); x11 = fmaxf(x11, 0.f);
        }
        C[(size_t)m0*N + n0] = (_Float16)x00;
        C[(size_t)m0*N + n1] = (_Float16)x01;
        C[(size_t)m1*N + n0] = (_Float16)x10;
        C[(size_t)m1*N + n1] = (_Float16)x11;
    }
}

__global__ void k_gemm_f32(const _Float16* __restrict__ A, const _Float16* __restrict__ B,
                           const float* __restrict__ bias, float* __restrict__ C,
                           int mtiles2, int npairs, int K) {
    int lane = threadIdx.x & 31;
    int wid  = blockIdx.x * (blockDim.x >> 5) + (threadIdx.x >> 5);
    if (wid >= mtiles2 * npairs) return;
    int mt = wid / npairs, np = wid % npairs;
    int lm = lane & 15, off = (lane >> 4) * 8;
    size_t arow0 = (size_t)(mt*32 + lm) * K;
    size_t arow1 = (size_t)(mt*32 + 16 + lm) * K;
    size_t brow0 = (size_t)(np*32 + lm) * K;
    size_t brow1 = (size_t)(np*32 + 16 + lm) * K;
    v8f z = {};
    v8f acc00 = z, acc01 = z, acc10 = z, acc11 = z;
    for (int k0 = 0; k0 < K; k0 += 32) {
        V16 a0, a1, b0, b1;
        load_frag(A + arow0 + k0 + off, a0);
        load_frag(A + arow1 + k0 + off, a1);
        load_frag(B + brow0 + k0 + off, b0);
        load_frag(B + brow1 + k0 + off, b1);
        acc00 = __builtin_amdgcn_wmma_f32_16x16x32_f16(false, a0.v, false, b0.v, (short)0, acc00, false, false);
        acc01 = __builtin_amdgcn_wmma_f32_16x16x32_f16(false, a0.v, false, b1.v, (short)0, acc01, false, false);
        acc10 = __builtin_amdgcn_wmma_f32_16x16x32_f16(false, a1.v, false, b0.v, (short)0, acc10, false, false);
        acc11 = __builtin_amdgcn_wmma_f32_16x16x32_f16(false, a1.v, false, b1.v, (short)0, acc11, false, false);
    }
    int n0 = np*32 + lm, n1 = n0 + 16;
    int N = npairs * 32;
    float bb0 = bias[n0], bb1 = bias[n1];
    #pragma unroll
    for (int v = 0; v < 8; ++v) {
        int m0 = mt*32 + v + 8*(lane >> 4);
        int m1 = m0 + 16;
        C[(size_t)m0*N + n0] = acc00[v] + bb0;
        C[(size_t)m0*N + n1] = acc01[v] + bb1;
        C[(size_t)m1*N + n0] = acc10[v] + bb0;
        C[(size_t)m1*N + n1] = acc11[v] + bb1;
    }
}

// ---------------- adaptive per-patch apply ----------------------------------
// out[l,f] = relu( sum_k xu[l,k] * kern[p, f*27+k] + bias[p,f] ) -> sp HWC f16
__global__ void k_adaptive(const float* __restrict__ x, const _Float16* __restrict__ kern,
                           const float* __restrict__ biasN, _Float16* __restrict__ sp) {
    __shared__ float   xs[3*7*7];
    __shared__ _Float16 ks[864];
    __shared__ float   bs[32];
    int p = blockIdx.x;
    int b = p / (HT*WT); int rem = p % (HT*WT);
    int ty = rem / WT, tx = rem % WT;
    for (int i = threadIdx.x; i < 147; i += blockDim.x) {
        int c = i / 49, r = (i % 49) / 7, q = i % 7;
        int pr = r - 1, pq = q - 1;                 // unfold pads the PATCH with zeros
        float v = 0.f;
        if (pr >= 0 && pr < 5 && pq >= 0 && pq < 5)
            v = x[(((size_t)b*3 + c)*HB + ty*5 + pr)*WB + tx*5 + pq];
        xs[i] = v;
    }
    for (int i = threadIdx.x; i < 864; i += blockDim.x) ks[i] = kern[(size_t)p*864 + i];
    for (int i = threadIdx.x; i < 32;  i += blockDim.x) bs[i] = biasN[(size_t)p*32 + i];
    __syncthreads();
    for (int item = threadIdx.x; item < 800; item += blockDim.x) {
        int f = item & 31, l = item >> 5;
        int py = l / 5, px = l % 5;
        float sum = bs[f];
        #pragma unroll
        for (int c = 0; c < 3; ++c)
            #pragma unroll
            for (int kh = 0; kh < 3; ++kh)
                #pragma unroll
                for (int kw = 0; kw < 3; ++kw)
                    sum += xs[c*49 + (py+kh)*7 + (px+kw)] * (float)ks[f*27 + c*9 + kh*3 + kw];
        sum = fmaxf(sum, 0.f);
        sp[(((size_t)b*HB + ty*5 + py)*WB + tx*5 + px)*32 + f] = (_Float16)sum;
    }
}

// ---------------- implicit-GEMM conv kernels --------------------------------
// conv1: 5x5 SAME, 32 -> 128, relu.  One wave = 16-pixel strip x 128 outputs.
__global__ void k_conv1(const _Float16* __restrict__ sp, const _Float16* __restrict__ W1,
                        const float* __restrict__ b1, _Float16* __restrict__ h1) {
    extern __shared__ char smem[];
    _Float16* lw = (_Float16*)smem;
    const int nvec = (25*8*32*16) / 8;
    for (int i = threadIdx.x; i < nvec; i += blockDim.x)
        ((uint4*)lw)[i] = ((const uint4*)W1)[i];
    __syncthreads();
    int lane  = threadIdx.x & 31;
    int strip = blockIdx.x * (blockDim.x >> 5) + (threadIdx.x >> 5);
    int b = strip / (HB*XSTRIPS); int rem = strip % (HB*XSTRIPS);
    int y = rem / XSTRIPS; int x0 = (rem % XSTRIPS) * 16;
    int lm = lane & 15, off = (lane >> 4) * 8;
    v8f z = {}; v8h hz = {};
    v8f acc[8];
    #pragma unroll
    for (int j = 0; j < 8; ++j) acc[j] = z;
    for (int kh = 0; kh < 5; ++kh) {
        int yy = y + kh - 2;
        if (yy < 0 || yy >= HB) continue;           // uniform per wave
        #pragma unroll
        for (int kw = 0; kw < 5; ++kw) {
            int t = kh*5 + kw;
            int xx = x0 + lm + kw - 2;              // per-lane row of A tile
            V16 a;
            if (xx >= 0 && xx < WB) {
                load_frag(sp + ((((size_t)b*HB + yy)*WB + xx)*32 + off), a);
            } else { a.h[0] = hz; a.h[1] = hz; }
            #pragma unroll
            for (int j = 0; j < 8; ++j) {
                v16h bf = *(const v16h*)(lw + (size_t)((t*8 + j)*32 + lane) * 16);
                acc[j] = __builtin_amdgcn_wmma_f32_16x16x32_f16(false, a.v, false, bf, (short)0, acc[j], false, false);
            }
        }
    }
    size_t pixbase = ((size_t)b*HB + y)*WB + x0;
    #pragma unroll
    for (int j = 0; j < 8; ++j) {
        int ch = j*16 + lm;
        float bb = b1[ch];
        #pragma unroll
        for (int v = 0; v < 8; ++v) {
            int m = v + 8*(lane >> 4);
            float val = fmaxf(acc[j][v] + bb, 0.f);
            h1[(pixbase + m)*128 + ch] = (_Float16)val;
        }
    }
}

// conv2: 3x3 SAME, 128 -> 32, relu. K = 4 chunks of 32 per tap.
__global__ void k_conv2(const _Float16* __restrict__ h1, const _Float16* __restrict__ W2,
                        const float* __restrict__ b2, _Float16* __restrict__ h2) {
    extern __shared__ char smem[];
    _Float16* lw = (_Float16*)smem;
    const int nvec = (9*4*2*32*16) / 8;
    for (int i = threadIdx.x; i < nvec; i += blockDim.x)
        ((uint4*)lw)[i] = ((const uint4*)W2)[i];
    __syncthreads();
    int lane  = threadIdx.x & 31;
    int strip = blockIdx.x * (blockDim.x >> 5) + (threadIdx.x >> 5);
    int b = strip / (HB*XSTRIPS); int rem = strip % (HB*XSTRIPS);
    int y = rem / XSTRIPS; int x0 = (rem % XSTRIPS) * 16;
    int lm = lane & 15, off = (lane >> 4) * 8;
    v8f z = {}; v8h hz = {};
    v8f acc[2]; acc[0] = z; acc[1] = z;
    for (int kh = 0; kh < 3; ++kh) {
        int yy = y + kh - 1;
        if (yy < 0 || yy >= HB) continue;
        #pragma unroll
        for (int kw = 0; kw < 3; ++kw) {
            int t = kh*3 + kw;
            int xx = x0 + lm + kw - 1;
            bool inb = (xx >= 0 && xx < WB);
            #pragma unroll
            for (int kc = 0; kc < 4; ++kc) {
                V16 a;
                if (inb) {
                    load_frag(h1 + ((((size_t)b*HB + yy)*WB + xx)*128 + kc*32 + off), a);
                } else { a.h[0] = hz; a.h[1] = hz; }
                #pragma unroll
                for (int j = 0; j < 2; ++j) {
                    v16h bf = *(const v16h*)(lw + (size_t)((((t*4 + kc)*2 + j)*32 + lane)) * 16);
                    acc[j] = __builtin_amdgcn_wmma_f32_16x16x32_f16(false, a.v, false, bf, (short)0, acc[j], false, false);
                }
            }
        }
    }
    size_t pixbase = ((size_t)b*HB + y)*WB + x0;
    #pragma unroll
    for (int j = 0; j < 2; ++j) {
        int ch = j*16 + lm;
        float bb = b2[ch];
        #pragma unroll
        for (int v = 0; v < 8; ++v) {
            int m = v + 8*(lane >> 4);
            float val = fmaxf(acc[j][v] + bb, 0.f);
            h2[(pixbase + m)*32 + ch] = (_Float16)val;
        }
    }
}

// conv3: 3x3 SAME, 32 -> 48 (= 3*4*4), no relu, fused pixel-shuffle to d_out f32
__global__ void k_conv3(const _Float16* __restrict__ h2, const _Float16* __restrict__ W3,
                        const float* __restrict__ b3, float* __restrict__ out) {
    extern __shared__ char smem[];
    _Float16* lw = (_Float16*)smem;
    const int nvec = (9*3*32*16) / 8;
    for (int i = threadIdx.x; i < nvec; i += blockDim.x)
        ((uint4*)lw)[i] = ((const uint4*)W3)[i];
    __syncthreads();
    int lane  = threadIdx.x & 31;
    int strip = blockIdx.x * (blockDim.x >> 5) + (threadIdx.x >> 5);
    int b = strip / (HB*XSTRIPS); int rem = strip % (HB*XSTRIPS);
    int y = rem / XSTRIPS; int x0 = (rem % XSTRIPS) * 16;
    int lm = lane & 15, off = (lane >> 4) * 8;
    v8f z = {}; v8h hz = {};
    v8f acc[3]; acc[0] = z; acc[1] = z; acc[2] = z;
    for (int kh = 0; kh < 3; ++kh) {
        int yy = y + kh - 1;
        if (yy < 0 || yy >= HB) continue;
        #pragma unroll
        for (int kw = 0; kw < 3; ++kw) {
            int t = kh*3 + kw;
            int xx = x0 + lm + kw - 1;
            V16 a;
            if (xx >= 0 && xx < WB) {
                load_frag(h2 + ((((size_t)b*HB + yy)*WB + xx)*32 + off), a);
            } else { a.h[0] = hz; a.h[1] = hz; }
            #pragma unroll
            for (int j = 0; j < 3; ++j) {
                v16h bf = *(const v16h*)(lw + (size_t)((t*3 + j)*32 + lane) * 16);
                acc[j] = __builtin_amdgcn_wmma_f32_16x16x32_f16(false, a.v, false, bf, (short)0, acc[j], false, false);
            }
        }
    }
    #pragma unroll
    for (int j = 0; j < 3; ++j) {
        int o = j*16 + lm;                   // o = c*16 + i*4 + jj
        float bb = b3[o];
        int c = o >> 4, ii = (o >> 2) & 3, jj = o & 3;
        #pragma unroll
        for (int v = 0; v < 8; ++v) {
            int m = v + 8*(lane >> 4);
            int px = x0 + m;
            out[(((size_t)b*3 + c)*(HB*4) + (y*4 + ii))*(size_t)(WB*4) + (px*4 + jj)] = acc[j][v] + bb;
        }
    }
}

// ---------------- host side --------------------------------------------------
extern "C" void kernel_launch(void* const* d_in, const int* in_sizes, int n_in,
                              void* d_out, int out_size, void* d_ws, size_t ws_size,
                              hipStream_t stream) {
    (void)in_sizes; (void)n_in; (void)out_size; (void)ws_size;
    const float* x      = (const float*)d_in[0];
    const float* w_feat = (const float*)d_in[1];
    const float* b_feat = (const float*)d_in[2];
    const float* w_kern = (const float*)d_in[3];
    const float* b_kern = (const float*)d_in[4];
    const float* w_badp = (const float*)d_in[5];
    const float* b_badp = (const float*)d_in[6];
    const float* w1 = (const float*)d_in[7];
    const float* b1 = (const float*)d_in[8];
    const float* w2 = (const float*)d_in[9];
    const float* b2 = (const float*)d_in[10];
    const float* w3 = (const float*)d_in[11];
    const float* b3 = (const float*)d_in[12];

    char* ws = (char*)d_ws;
    size_t off = 0;
    auto alloc = [&](size_t bytes) -> void* {
        void* p = ws + off;
        off += (bytes + 255) & ~(size_t)255;
        return p;
    };
    _Float16* A1   = (_Float16*)alloc((size_t)NPATCH*9*32*2);      // 10.6 MB
    _Float16* feat = (_Float16*)alloc((size_t)NPATCH*9*256*2);     // 85 MB (== A2 18432x2304)
    _Float16* kern = (_Float16*)alloc((size_t)NPATCH*864*2);       // 32 MB
    float*    bN   = (float*)   alloc((size_t)NPATCH*32*4);        // 2.4 MB
    _Float16* sp   = (_Float16*)alloc((size_t)2*HB*WB*32*2);       // 29 MB
    _Float16* h1   = (_Float16*)alloc((size_t)2*HB*WB*128*2);      // 118 MB
    _Float16* h2   = (_Float16*)alloc((size_t)2*HB*WB*32*2);       // 29 MB
    _Float16* B1   = (_Float16*)alloc((size_t)256*32*2);
    _Float16* B2   = (_Float16*)alloc((size_t)864*2304*2);
    _Float16* B3   = (_Float16*)alloc((size_t)32*2304*2);
    _Float16* W1p  = (_Float16*)alloc((size_t)25*8*32*16*2);
    _Float16* W2p  = (_Float16*)alloc((size_t)9*4*2*32*16*2);
    _Float16* W3p  = (_Float16*)alloc((size_t)9*3*32*16*2);

    // weight packs + im2col (independent)
    k_im2col1<<<(NPATCH*9 + 255)/256, 256, 0, stream>>>(x, A1);
    k_packB1 <<<32,   256, 0, stream>>>(w_feat, B1);
    k_packB29<<<7776, 256, 0, stream>>>(w_kern, B2, 864);
    k_packB29<<<288,  256, 0, stream>>>(w_badp, B3, 32);
    k_packW1 <<<400,  256, 0, stream>>>(w1, W1p);
    k_packW2 <<<144,  256, 0, stream>>>(w2, W2p);
    k_packW3 <<<54,   256, 0, stream>>>(w3, W3p);

    // feat = relu(A1 x B1^T + b_feat): M=165888 (5184 m-pair tiles), N=256 (8), K=32
    // waves = 5184*8 = 41472 -> 5184 blocks of 8 waves
    k_gemm_f16<<<5184, 256, 0, stream>>>(A1, B1, b_feat, feat, 5184, 8, 32, 1);
    // kern = feat x B2^T + b_kern: M=18432 (576), N=864 (27), K=2304 -> 15552 waves
    k_gemm_f16<<<1944, 256, 0, stream>>>(feat, B2, b_kern, kern, 576, 27, 2304, 0);
    // bias = feat x B3^T + b_badp: N=32 (1 pair) -> 576 waves
    k_gemm_f32<<<72, 256, 0, stream>>>(feat, B3, b_badp, bN, 576, 1, 2304);
    // adaptive apply + batch_to_space -> sp (B,H,W,32) f16
    k_adaptive<<<NPATCH, 128, 0, stream>>>(x, kern, bN, sp);

    // regular conv stack (implicit GEMM, weights staged in LDS)
    k_conv1<<<NSTRIP/8, 256, 25*8*32*16*2, stream>>>(sp, W1p, b1, h1);
    k_conv2<<<NSTRIP/8, 256, 9*4*2*32*16*2, stream>>>(h1, W2p, b2, h2);
    k_conv3<<<NSTRIP/8, 256, 9*3*32*16*2, stream>>>(h2, W3p, b3, (float*)d_out);
}
